// MOEViTCurveAdapter_37933151158769
// MI455X (gfx1250) — matrix operand order repaired and checked by reading, not compiled, not run on hardware
//
#include <hip/hip_runtime.h>

#define B_N 8192
#define E_N 8
#define K_N 2
#define D_N 3072
#define H_N 256
#define C_N 100
#define CP  112            // C padded to 7*16
#define MT  128            // rows per GEMM tile
#define KS1 (D_N / 32)     // 96 k-steps, GEMM1
#define NT1 (H_N / 16)     // 16 n-tiles, GEMM1
#define KS2 (H_N / 32)     // 8 k-steps, GEMM2
#define NT2 (CP  / 16)     // 7 n-tiles, GEMM2
#define SLAB 16384         // one W1 k-slab in LDS (16 frags x 1 KB)

typedef __attribute__((ext_vector_type(16))) __bf16 v16bf;
typedef __attribute__((ext_vector_type(8)))  float  v8f;
typedef __attribute__((ext_vector_type(4)))  unsigned u32x4;

union Frag { u32x4 q[2]; v16bf v; };

__device__ __forceinline__ unsigned short f2bf(float f) {
    union { float f; unsigned u; } a; a.f = f;
    unsigned r = a.u + 0x7FFFu + ((a.u >> 16) & 1u);   // round-to-nearest-even
    return (unsigned short)(r >> 16);
}

// ---------------------------------------------------------------------------
// Kernel 1: small fp32 blends (router W/b, expert biases; B2 padded to 112)
// ---------------------------------------------------------------------------
__global__ __launch_bounds__(256) void blend_small(
    const float* __restrict__ co, const float* __restrict__ rw,
    const float* __restrict__ rb, const float* __restrict__ b1,
    const float* __restrict__ b2,
    float* __restrict__ Wr, float* __restrict__ br,
    float* __restrict__ B1b, float* __restrict__ B2b)
{
    int t = blockIdx.x * 256 + threadIdx.x;
    float c0 = co[0], c1 = co[1];
    if (t < D_N * E_N) { Wr[t] = c0 * rw[t] + c1 * rw[D_N * E_N + t]; return; }
    t -= D_N * E_N;
    if (t < E_N) { br[t] = c0 * rb[t] + c1 * rb[E_N + t]; return; }
    t -= E_N;
    if (t < E_N * H_N) {
        int e = t / H_N, h = t % H_N;
        B1b[t] = c0 * b1[(e * K_N + 0) * H_N + h] + c1 * b1[(e * K_N + 1) * H_N + h];
        return;
    }
    t -= E_N * H_N;
    if (t < E_N * CP) {
        int e = t / CP, c = t % CP;
        float v = 0.f;
        if (c < C_N) v = c0 * b2[(e * K_N + 0) * C_N + c] + c1 * b2[(e * K_N + 1) * C_N + c];
        B2b[t] = v;
    }
}

// ---------------------------------------------------------------------------
// Kernel 2: blend W1 -> bf16, pre-swizzled into WMMA B-fragment order.
// B layout (16-bit, 32x16): lane<16 -> col n=lane%16, K=0..7,16..23 ; lane>=16 -> K=8..15,24..31.
// ---------------------------------------------------------------------------
__global__ __launch_bounds__(256) void blend_w1s(
    const float* __restrict__ co, const float* __restrict__ w1,
    unsigned* __restrict__ W1s)
{
    int p = blockIdx.x * 256 + threadIdx.x;           // one dword (2 elems)
    float c0 = co[0], c1 = co[1];
    int j    = (p & 7) * 2;
    int lane = (p >> 3) & 31;
    int f    = p >> 8;
    int ks   = f % KS1;
    int nt   = (f / KS1) % NT1;
    int e    =  f / (KS1 * NT1);
    int kl   = ((lane < 16) ? 0 : 8) + ((j < 8) ? j : j + 8);
    int d    = ks * 32 + kl;                          // and d+1
    int h    = nt * 16 + (lane & 15);
    size_t i0 = ((size_t)(e * K_N + 0) * D_N + d) * H_N + h;
    size_t i1 = ((size_t)(e * K_N + 1) * D_N + d) * H_N + h;
    float v0 = c0 * w1[i0]       + c1 * w1[i1];
    float v1 = c0 * w1[i0 + H_N] + c1 * w1[i1 + H_N];
    W1s[p] = (unsigned)f2bf(v0) | ((unsigned)f2bf(v1) << 16);
}

// ---------------------------------------------------------------------------
// Kernel 3: blend W2 -> bf16, same B-fragment swizzle, cols >= 100 zero-padded.
// ---------------------------------------------------------------------------
__global__ __launch_bounds__(256) void blend_w2s(
    const float* __restrict__ co, const float* __restrict__ w2,
    unsigned* __restrict__ W2s)
{
    int p = blockIdx.x * 256 + threadIdx.x;
    float c0 = co[0], c1 = co[1];
    int j    = (p & 7) * 2;
    int lane = (p >> 3) & 31;
    int f    = p >> 8;
    int ks   = f % KS2;
    int nt   = (f / KS2) % NT2;
    int e    =  f / (KS2 * NT2);
    int kl   = ((lane < 16) ? 0 : 8) + ((j < 8) ? j : j + 8);
    int hh   = ks * 32 + kl;
    int n    = nt * 16 + (lane & 15);
    unsigned outw = 0;
    if (n < C_N) {
        size_t i0 = ((size_t)(e * K_N + 0) * H_N + hh) * C_N + n;
        size_t i1 = ((size_t)(e * K_N + 1) * H_N + hh) * C_N + n;
        float v0 = c0 * w2[i0]       + c1 * w2[i1];
        float v1 = c0 * w2[i0 + C_N] + c1 * w2[i1 + C_N];
        outw = (unsigned)f2bf(v0) | ((unsigned)f2bf(v1) << 16);
    }
    W2s[p] = outw;
}

// ---------------------------------------------------------------------------
// Kernel 4: router (fp32) + x->bf16 conversion + expert bucketing.
// One wave32 per sample.
// ---------------------------------------------------------------------------
__global__ __launch_bounds__(256) void router_kernel(
    const float* __restrict__ x, const float* __restrict__ Wr,
    const float* __restrict__ br, unsigned short* __restrict__ xbf,
    int* __restrict__ cnt, int* __restrict__ list)
{
    int wave = threadIdx.x >> 5;
    int lane = threadIdx.x & 31;
    int b = blockIdx.x * 8 + wave;

    float acc[E_N];
#pragma unroll
    for (int e = 0; e < E_N; ++e) acc[e] = 0.f;

    const float* xr = x + (size_t)b * D_N;
    for (int d = lane; d < D_N; d += 32) {
        float xv = xr[d];
        xbf[(size_t)b * D_N + d] = f2bf(xv);
        const float* wr = Wr + (size_t)d * E_N;
#pragma unroll
        for (int e = 0; e < E_N; ++e) acc[e] += xv * wr[e];
    }
#pragma unroll
    for (int off = 16; off > 0; off >>= 1)
#pragma unroll
        for (int e = 0; e < E_N; ++e) acc[e] += __shfl_xor(acc[e], off, 32);

    if (lane == 0) {
        int best = 0; float bv = acc[0] + br[0];
#pragma unroll
        for (int e = 1; e < E_N; ++e) {
            float v = acc[e] + br[e];
            if (v > bv) { bv = v; best = e; }       // first-max wins (argmax tie rule)
        }
        int pos = atomicAdd(&cnt[best], 1);
        list[best * B_N + pos] = b;
    }
}

// ---------------------------------------------------------------------------
// Kernel 5: fused MoE MLP over dispatched rows.
// GEMM1: W1 k-slabs streamed into double-buffered 2x16KB LDS via
//        global_load_async_to_lds_b128 (ASYNCcnt), one barrier per k-step.
//        sched_group_barrier enforces a DS-read/WMMA software pipeline
//        (each WMMA overlaps the loads of fragment nt+2).
// Epilogue: +bias, exact gelu, bf16 h -> LDS (64 KB, overlaps the slabs).
// GEMM2: [128x256]x[256x112] bf16 WMMA, bias preloaded, scatter to out.
// ---------------------------------------------------------------------------
__global__ __launch_bounds__(256) void moe_gemm(
    const unsigned short* __restrict__ xbf,
    const u32x4* __restrict__ W1s, const u32x4* __restrict__ W2s,
    const float* __restrict__ B1b, const float* __restrict__ B2b,
    const int* __restrict__ cnt, const int* __restrict__ list,
    float* __restrict__ out)
{
    __shared__ __align__(16) unsigned short h_lds[MT][H_N];        // 64 KB
    char* lds_raw = (char*)&h_lds[0][0];                           // slabs overlap h

    int e    = blockIdx.x >> 6;        // 64 tiles per expert
    int tile = blockIdx.x & 63;
    int m0   = tile * MT;
    int count = cnt[e];
    if (m0 >= count) return;
    int rows_valid = count - m0; if (rows_valid > MT) rows_valid = MT;

    int tid = threadIdx.x, wave = tid >> 5, lane = tid & 31;
    int lrow = wave * 16 + (lane & 15);
    int srcRow = list[e * B_N + m0 + ((lrow < rows_valid) ? lrow : 0)];
    const unsigned short* arow = xbf + (size_t)srcRow * D_N;
    int abase = (lane < 16) ? 0 : 8;   // 16-bit A layout K split

    v8f acc[NT1];
#pragma unroll
    for (int i = 0; i < NT1; ++i)
#pragma unroll
        for (int r = 0; r < 8; ++r) acc[i][r] = 0.f;

    const u32x4* w1e = W1s + (size_t)e * (NT1 * KS1 * 64);
    const unsigned lds_base = (unsigned)(uintptr_t)lds_raw;

    // async fill of slab for k-step ks into buffer (ks & 1)
    auto fill_slab = [&](int ks) {
        int nt = tid >> 4, sub = tid & 15;                       // 64 B per thread
        const u32x4* s = w1e + ((size_t)nt * KS1 + ks) * 64 + sub * 4;
        unsigned dlds = lds_base + (unsigned)((ks & 1) * SLAB + (nt * 64 + sub * 4) * 16);
        unsigned long long ga = (unsigned long long)(uintptr_t)s;
        asm volatile(
            "global_load_async_to_lds_b128 %0, %1, off\n\t"
            "global_load_async_to_lds_b128 %0, %1, off offset:16\n\t"
            "global_load_async_to_lds_b128 %0, %1, off offset:32\n\t"
            "global_load_async_to_lds_b128 %0, %1, off offset:48"
            :: "v"(dlds), "v"(ga) : "memory");
    };
    auto load_a = [&](int ks, Frag& a) {
        const u32x4* ap = (const u32x4*)(arow + ks * 32 + abase);
        a.q[0] = ap[0];            // K = abase+0..7
        a.q[1] = ap[2];            // K = abase+16..23
    };

    Frag a0, a1;
    fill_slab(0);
    load_a(0, a0);

    auto body = [&](int ks, Frag& acur, Frag& anext, unsigned bufoff) {
        asm volatile("s_wait_asynccnt 0x0" ::: "memory");  // my slab stores done
        __syncthreads();                                   // slab ks published
        if (ks + 1 < KS1) { fill_slab(ks + 1); load_a(ks + 1, anext); }
        const u32x4* cur = (const u32x4*)(lds_raw + bufoff);
        Frag b0, b1;
        b0.q[0] = cur[lane * 2 + 0];
        b0.q[1] = cur[lane * 2 + 1];
#pragma unroll
        for (int nt = 0; nt < NT1; ++nt) {
            Frag* bc = (nt & 1) ? &b1 : &b0;
            Frag* bn = (nt & 1) ? &b0 : &b1;
            if (nt + 1 < NT1) {
                bn->q[0] = cur[(nt + 1) * 64 + lane * 2 + 0];
                bn->q[1] = cur[(nt + 1) * 64 + lane * 2 + 1];
            }
            acc[nt] = __builtin_amdgcn_wmma_f32_16x16x32_bf16(
                false, acur.v, false, bc->v, (short)0, acc[nt], false, false);
        }
#if defined(__has_builtin)
#if __has_builtin(__builtin_amdgcn_sched_group_barrier)
        // Pipeline directive: 32 DS reads + 16 WMMAs per k-step.
        // Order: DS(4) [frags 0,1], then 14x { WMMA(1), DS(2) [frag nt+2] },
        // then the last two WMMAs. Each WMMA overlaps a 2-fragment load slack.
        __builtin_amdgcn_sched_group_barrier(0x100, 4, 0);   // DS read
#pragma unroll
        for (int i = 0; i < 14; ++i) {
            __builtin_amdgcn_sched_group_barrier(0x008, 1, 0);  // WMMA
            __builtin_amdgcn_sched_group_barrier(0x100, 2, 0);  // DS read
        }
        __builtin_amdgcn_sched_group_barrier(0x008, 1, 0);
        __builtin_amdgcn_sched_group_barrier(0x008, 1, 0);
#endif
#endif
    };

    for (int ks = 0; ks < KS1; ks += 2) {
        body(ks,     a0, a1, 0);
        body(ks + 1, a1, a0, SLAB);
    }
    __syncthreads();

    // epilogue 1: bias + exact gelu -> bf16 h in LDS
    {
        int hiHalf = lane >> 4, ncol = lane & 15;
#pragma unroll
        for (int nt = 0; nt < NT1; ++nt) {
            int n = nt * 16 + ncol;
            float bias = B1b[e * H_N + n];
#pragma unroll
            for (int r = 0; r < 8; ++r) {
                int m = wave * 16 + r + hiHalf * 8;
                float v = acc[nt][r] + bias;
                v = 0.5f * v * (1.0f + erff(v * 0.70710678118654752f));
                h_lds[m][n] = f2bf(v);
            }
        }
    }
    __syncthreads();

    // GEMM2: h[128x256] @ W2[256x112], bias preloaded into accumulators
    v8f acc2[NT2];
    {
        int ncol = lane & 15;
#pragma unroll
        for (int nt = 0; nt < NT2; ++nt) {
            float bv = B2b[e * CP + nt * 16 + ncol];
#pragma unroll
            for (int r = 0; r < 8; ++r) acc2[nt][r] = bv;
        }
    }
    const unsigned short* hrow = &h_lds[wave * 16 + (lane & 15)][0];
    const u32x4* w2e = W2s + (size_t)e * (NT2 * KS2 * 64);
    for (int ks = 0; ks < KS2; ++ks) {
        Frag a;
        const u32x4* ap = (const u32x4*)(hrow + ks * 32 + abase);
        a.q[0] = ap[0];
        a.q[1] = ap[2];
        Frag b0, b1;
        const u32x4* bp0 = w2e + ((size_t)0 * KS2 + ks) * 64 + lane * 2;
        b0.q[0] = bp0[0];
        b0.q[1] = bp0[1];
#pragma unroll
        for (int nt = 0; nt < NT2; ++nt) {
            Frag* bc = (nt & 1) ? &b1 : &b0;
            Frag* bn = (nt & 1) ? &b0 : &b1;
            if (nt + 1 < NT2) {
                const u32x4* bp = w2e + ((size_t)(nt + 1) * KS2 + ks) * 64 + lane * 2;
                bn->q[0] = bp[0];
                bn->q[1] = bp[1];
            }
            acc2[nt] = __builtin_amdgcn_wmma_f32_16x16x32_bf16(
                false, a.v, false, bc->v, (short)0, acc2[nt], false, false);
        }
    }

    // epilogue 2: scatter valid rows to d_out
    {
        int hiHalf = lane >> 4, ncol = lane & 15;
#pragma unroll
        for (int r = 0; r < 8; ++r) {
            int m = wave * 16 + r + hiHalf * 8;
            if (m < rows_valid) {
                int g = list[e * B_N + m0 + m];
                float* orow = out + (size_t)g * C_N;
#pragma unroll
                for (int nt = 0; nt < NT2; ++nt) {
                    int n = nt * 16 + ncol;
                    if (n < C_N) orow[n] = acc2[nt][r];
                }
            }
        }
    }
}

// ---------------------------------------------------------------------------
extern "C" void kernel_launch(void* const* d_in, const int* in_sizes, int n_in,
                              void* d_out, int out_size, void* d_ws, size_t ws_size,
                              hipStream_t stream) {
    (void)in_sizes; (void)n_in; (void)out_size; (void)ws_size;
    const float* x  = (const float*)d_in[0];
    const float* co = (const float*)d_in[1];
    const float* rw = (const float*)d_in[2];
    const float* rb = (const float*)d_in[3];
    const float* w1 = (const float*)d_in[4];
    const float* b1 = (const float*)d_in[5];
    const float* w2 = (const float*)d_in[6];
    const float* b2 = (const float*)d_in[7];
    float* out = (float*)d_out;

    char* ws = (char*)d_ws;
    size_t off = 0;
    auto alloc = [&](size_t bytes) -> char* {
        char* p = ws + off;
        off += (bytes + 255) & ~(size_t)255;
        return p;
    };
    unsigned short* xbf = (unsigned short*)alloc((size_t)B_N * D_N * 2);          // 48 MB
    unsigned*       W1s = (unsigned*)      alloc((size_t)E_N * NT1 * KS1 * 1024); // 12.6 MB
    unsigned*       W2s = (unsigned*)      alloc((size_t)E_N * NT2 * KS2 * 1024); // 0.46 MB
    float*          Wr  = (float*)         alloc((size_t)D_N * E_N * 4);
    float*          br  = (float*)         alloc(E_N * 4);
    float*          B1b = (float*)         alloc((size_t)E_N * H_N * 4);
    float*          B2b = (float*)         alloc((size_t)E_N * CP * 4);
    int*            cnt = (int*)           alloc(E_N * 4);
    int*            lst = (int*)           alloc((size_t)E_N * B_N * 4);

    hipMemsetAsync(cnt, 0, E_N * sizeof(int), stream);

    blend_small<<<dim3(108), dim3(256), 0, stream>>>(co, rw, rb, b1, b2, Wr, br, B1b, B2b);
    blend_w1s<<<dim3((E_N * NT1 * KS1 * 256) / 256), dim3(256), 0, stream>>>(co, w1, W1s);
    blend_w2s<<<dim3((E_N * NT2 * KS2 * 256) / 256), dim3(256), 0, stream>>>(co, w2, W2s);
    router_kernel<<<dim3(B_N / 8), dim3(256), 0, stream>>>(x, Wr, br, xbf, cnt, lst);
    moe_gemm<<<dim3(E_N * (B_N / MT)), dim3(256), 0, stream>>>(
        xbf, (const u32x4*)W1s, (const u32x4*)W2s, B1b, B2b, cnt, lst, out);
}
